// OptimizedGeometryAwareCrossAttention_16939351015986
// MI455X (gfx1250) — matrix operand order repaired and checked by reading, not compiled, not run on hardware
//
#include <hip/hip_runtime.h>
#include <hip/hip_bf16.h>

typedef unsigned short u16;
typedef __attribute__((ext_vector_type(16))) __bf16 v16bf;
typedef __attribute__((ext_vector_type(8)))  float  v8f;

#define WAVE 32

// ---------- helpers ----------
__device__ __forceinline__ u16 f2bf(float f) {
    unsigned int u = __float_as_uint(f);
    u += 0x7fffu + ((u >> 16) & 1u);          // round-to-nearest-even
    return (u16)(u >> 16);
}
__device__ __forceinline__ float wred_sum(float v) {
#pragma unroll
    for (int m = 16; m >= 1; m >>= 1) v += __shfl_xor(v, m, WAVE);
    return v;
}
__device__ __forceinline__ float wred_max(float v) {
#pragma unroll
    for (int m = 16; m >= 1; m >>= 1) v = fmaxf(v, __shfl_xor(v, m, WAVE));
    return v;
}

// ---------- block boundaries from sorted block_id ----------
__global__ void k_boundaries(const int* __restrict__ bid, int* __restrict__ start,
                             int* __restrict__ end, int N) {
    int i = blockIdx.x * blockDim.x + threadIdx.x;
    if (i >= N) return;
    int b = bid[i];
    if (i == 0 || bid[i - 1] != b) start[b] = i;
    if (i == N - 1 || bid[i + 1] != b) end[b] = i + 1;
}

// ---------- per-block centroid (one wave per block) ----------
__global__ void k_centroid(const float* __restrict__ pos, const int* __restrict__ start,
                           const int* __restrict__ end, float* __restrict__ cent, int B) {
    int b = blockIdx.x * (blockDim.x >> 5) + (threadIdx.x >> 5);
    int lane = threadIdx.x & 31;
    if (b >= B) return;
    int s = start[b], e = end[b];
    float sx = 0.f, sy = 0.f, sz = 0.f;
    for (int i = s + lane; i < e; i += WAVE) {
        sx += pos[(size_t)i * 3 + 0];
        sy += pos[(size_t)i * 3 + 1];
        sz += pos[(size_t)i * 3 + 2];
    }
    sx = wred_sum(sx); sy = wred_sum(sy); sz = wred_sum(sz);
    if (lane == 0) {
        float cnt = fmaxf((float)(e - s), 1.0f);
        cent[(size_t)b * 3 + 0] = sx / cnt;
        cent[(size_t)b * 3 + 1] = sy / cnt;
        cent[(size_t)b * 3 + 2] = sz / cnt;
    }
}

// ---------- RBF geometry + build augmented bf16 rows [N, H+Hq] ----------
__global__ void k_geom_aug(const float* __restrict__ af, const float* __restrict__ pos,
                           const int* __restrict__ bid, const float* __restrict__ cent,
                           const float* __restrict__ centers, const float* __restrict__ widths,
                           const float* __restrict__ Wg, const float* __restrict__ bg,
                           u16* __restrict__ aug, int N, int H, int R, int Hq) {
    int i = blockIdx.x * (blockDim.x >> 5) + (threadIdx.x >> 5);
    int lane = threadIdx.x & 31;
    if (i >= N) return;
    int b = bid[i];
    float rx = pos[(size_t)i * 3 + 0] - cent[(size_t)b * 3 + 0];
    float ry = pos[(size_t)i * 3 + 1] - cent[(size_t)b * 3 + 1];
    float rz = pos[(size_t)i * 3 + 2] - cent[(size_t)b * 3 + 2];
    float dist = sqrtf(rx * rx + ry * ry + rz * rz);

    const float* frow = af + (size_t)i * H;
    u16* arow = aug + (size_t)i * (H + Hq);
#pragma unroll
    for (int t = 0; t < 8; ++t)
        arow[lane * 8 + t] = f2bf(frow[lane * 8 + t]);

    int j0 = lane * 2;          // Hq == 64 -> 2 outputs per lane
    float g0 = bg[j0], g1 = bg[j0 + 1];
    for (int r = 0; r < R; ++r) {
        float dr = dist - centers[r];
        float w  = widths[r];
        float e  = __expf(-(dr * dr) / (2.0f * w * w));
        g0 += e * Wg[r * Hq + j0];
        g1 += e * Wg[r * Hq + j0 + 1];
    }
    arow[H + j0]     = f2bf(g0);
    arow[H + j0 + 1] = f2bf(g1);
}

// ---------- convert f32 [K,N] weight -> bf16 transposed [N,K] ----------
__global__ void k_wt_transpose(const float* __restrict__ W, u16* __restrict__ Wt, int K, int N) {
    int idx = blockIdx.x * blockDim.x + threadIdx.x;
    if (idx >= K * N) return;
    int n = idx / K, k = idx - n * K;
    Wt[idx] = f2bf(W[(size_t)k * N + n]);
}

// ---------- elementwise f32 -> bf16 ----------
__global__ void k_f2bf(const float* __restrict__ in, u16* __restrict__ out, size_t n) {
    size_t i = (size_t)blockIdx.x * blockDim.x + threadIdx.x;
    if (i < n) out[i] = f2bf(in[i]);
}

// ---------- bf16 WMMA GEMM: out[M,Nd] = A[M,Kd] * Bt[Nd,Kd]^T + bias ----------
// one wave -> 16x64 strip (4 accumulators, A fragment reused across 4 WMMAs);
// block = 8 waves stacked along M -> 128x64 per workgroup
__device__ __forceinline__ v16bf ldfrag(const u16* p0, const u16* p1) {
    v16bf f;
    uint4* fp = (uint4*)&f;
    fp[0] = *(const uint4*)p0;
    fp[1] = *(const uint4*)p1;
    return f;
}

__global__ void k_gemm_bf16(const u16* __restrict__ A, const u16* __restrict__ Bt,
                            const float* __restrict__ bias,
                            float* __restrict__ outF, u16* __restrict__ outB,
                            int M, int Kd, int Nd, int relu) {
    int wave = threadIdx.x >> 5;
    int lane = threadIdx.x & 31;
    int m0 = (blockIdx.y * 8 + wave) * 16;
    int n0 = blockIdx.x * 64;
    if (m0 >= M) return;
    int half = lane >> 4;
    int l15  = lane & 15;

    const u16* arow = A  + (size_t)(m0 + l15) * Kd;
    const u16* br0  = Bt + (size_t)(n0 +  0 + l15) * Kd;
    const u16* br1  = Bt + (size_t)(n0 + 16 + l15) * Kd;
    const u16* br2  = Bt + (size_t)(n0 + 32 + l15) * Kd;
    const u16* br3  = Bt + (size_t)(n0 + 48 + l15) * Kd;

    v8f acc0 = {}, acc1 = {}, acc2 = {}, acc3 = {};
    for (int k0 = 0; k0 < Kd; k0 += 32) {
        // A fragment: 16 contiguous K starting at k0+half*16
        v16bf a = ldfrag(arow + k0 + half * 16, arow + k0 + half * 16 + 8);
        // B fragments: lane = N column, 16 contiguous K starting at k0+half*16
        int kb = k0 + half * 16;
        v16bf b0 = ldfrag(br0 + kb, br0 + kb + 8);
        v16bf b1 = ldfrag(br1 + kb, br1 + kb + 8);
        v16bf b2 = ldfrag(br2 + kb, br2 + kb + 8);
        v16bf b3 = ldfrag(br3 + kb, br3 + kb + 8);
        if (k0 + 64 < Kd) __builtin_prefetch(arow + k0 + 64, 0, 1);
        // reuse_a hint on the first three: the following identical WMMA reuses A
        acc0 = __builtin_amdgcn_wmma_f32_16x16x32_bf16(false, a, false, b0,
                                                       (short)0, acc0, true,  false);
        acc1 = __builtin_amdgcn_wmma_f32_16x16x32_bf16(false, a, false, b1,
                                                       (short)0, acc1, true,  false);
        acc2 = __builtin_amdgcn_wmma_f32_16x16x32_bf16(false, a, false, b2,
                                                       (short)0, acc2, true,  false);
        acc3 = __builtin_amdgcn_wmma_f32_16x16x32_bf16(false, a, false, b3,
                                                       (short)0, acc3, false, false);
    }

    v8f accs[4] = {acc0, acc1, acc2, acc3};
#pragma unroll
    for (int t = 0; t < 4; ++t) {
        int ncol = n0 + t * 16 + l15;
        float bv = bias[ncol];
#pragma unroll
        for (int v = 0; v < 8; ++v) {
            int row = m0 + half * 8 + v;
            float val = accs[t][v] + bv;
            if (relu) val = fmaxf(val, 0.0f);
            size_t idx = (size_t)row * Nd + ncol;
            if (outB) outB[idx] = f2bf(val);
            else      outF[idx] = val;
        }
    }
}

// ---------- logits: dot(Q[bid[i]], K[i]) / sqrt(H)  (one wave per atom) ----------
__global__ void k_logits(const float* __restrict__ Q, const float* __restrict__ Kmat,
                         const int* __restrict__ bid, float* __restrict__ logits,
                         int N, int H, float inv_sqrt_h) {
    int i = blockIdx.x * (blockDim.x >> 5) + (threadIdx.x >> 5);
    int lane = threadIdx.x & 31;
    if (i >= N) return;
    const float* q = Q + (size_t)bid[i] * H + lane * 8;
    const float* k = Kmat + (size_t)i * H + lane * 8;
    float s = 0.f;
#pragma unroll
    for (int t = 0; t < 8; ++t) s += q[t] * k[t];
    s = wred_sum(s);
    if (lane == 0) logits[i] = s * inv_sqrt_h;
}

// ---------- per-block softmax + context (one wave per block) ----------
__global__ void k_attention(const float* __restrict__ logits, const float* __restrict__ V,
                            const int* __restrict__ start, const int* __restrict__ end,
                            float* __restrict__ ctx, int B, int H) {
    int b = blockIdx.x * (blockDim.x >> 5) + (threadIdx.x >> 5);
    int lane = threadIdx.x & 31;
    if (b >= B) return;
    int s = start[b], e = end[b];
    float* crow = ctx + (size_t)b * H + lane * 8;
    if (e <= s) {
#pragma unroll
        for (int t = 0; t < 8; ++t) crow[t] = 0.f;
        return;
    }
    float m = -3.4028235e38f;
    for (int i = s + lane; i < e; i += WAVE) m = fmaxf(m, logits[i]);
    m = wred_max(m);
    float d = 0.f;
    for (int i = s + lane; i < e; i += WAVE) d += __expf(logits[i] - m);
    d = wred_sum(d);
    float inv = 1.0f / d;
    float acc[8] = {0.f, 0.f, 0.f, 0.f, 0.f, 0.f, 0.f, 0.f};
    for (int i = s; i < e; ++i) {
        float w = __expf(logits[i] - m) * inv;
        const float* vr = V + (size_t)i * H + lane * 8;
#pragma unroll
        for (int t = 0; t < 8; ++t) acc[t] += w * vr[t];
    }
#pragma unroll
    for (int t = 0; t < 8; ++t) crow[t] = acc[t];
}

// ---------- LayerNorm(res1 + res2) -> f32 (and optional bf16 copy) ----------
__global__ void k_add_ln(const float* __restrict__ a, const float* __restrict__ c,
                         const int* __restrict__ bid, const float* __restrict__ g,
                         const float* __restrict__ beta, float* __restrict__ outF,
                         u16* __restrict__ outB, int N, int H) {
    int i = blockIdx.x * (blockDim.x >> 5) + (threadIdx.x >> 5);
    int lane = threadIdx.x & 31;
    if (i >= N) return;
    size_t crow = bid ? (size_t)bid[i] * H : (size_t)i * H;
    float v[8];
    float s = 0.f;
#pragma unroll
    for (int t = 0; t < 8; ++t) {
        v[t] = a[(size_t)i * H + lane * 8 + t] + c[crow + lane * 8 + t];
        s += v[t];
    }
    float mean = wred_sum(s) * (1.0f / (float)H);
    float q = 0.f;
#pragma unroll
    for (int t = 0; t < 8; ++t) { float d = v[t] - mean; q += d * d; }
    float var = wred_sum(q) * (1.0f / (float)H);
    float rstd = rsqrtf(var + 1e-5f);
#pragma unroll
    for (int t = 0; t < 8; ++t) {
        int col = lane * 8 + t;
        float y = (v[t] - mean) * rstd * g[col] + beta[col];
        outF[(size_t)i * H + col] = y;
        if (outB) outB[(size_t)i * H + col] = f2bf(y);
    }
}

// ============================================================================
extern "C" void kernel_launch(void* const* d_in, const int* in_sizes, int n_in,
                              void* d_out, int out_size, void* d_ws, size_t ws_size,
                              hipStream_t stream) {
    const int H = 256, R = 16, Hq = 64;
    const int Haug = H + Hq;                  // 320
    const int N = in_sizes[0] / H;            // atoms
    const int B = in_sizes[2] / H;            // blocks
    const int F = in_sizes[19];               // 2*H = 512

    const float* af     = (const float*)d_in[0];
    const float* pos    = (const float*)d_in[1];
    const float* bfeat  = (const float*)d_in[2];
    const int*   bid    = (const int*)  d_in[3];
    const float* ctr    = (const float*)d_in[4];
    const float* wid    = (const float*)d_in[5];
    const float* Wg     = (const float*)d_in[6];
    const float* bg     = (const float*)d_in[7];
    const float* Wq     = (const float*)d_in[8];
    const float* bq     = (const float*)d_in[9];
    const float* Wk     = (const float*)d_in[10];
    const float* bk     = (const float*)d_in[11];
    const float* Wv     = (const float*)d_in[12];
    const float* bv     = (const float*)d_in[13];
    const float* Wc1    = (const float*)d_in[14];
    const float* bc1    = (const float*)d_in[15];
    const float* Wc2    = (const float*)d_in[16];
    const float* bc2    = (const float*)d_in[17];
    const float* Wf1    = (const float*)d_in[18];
    const float* bf1    = (const float*)d_in[19];
    const float* Wf2    = (const float*)d_in[20];
    const float* bf2    = (const float*)d_in[21];
    const float* g1     = (const float*)d_in[22];
    const float* b1     = (const float*)d_in[23];
    const float* g2     = (const float*)d_in[24];
    const float* b2     = (const float*)d_in[25];
    float* out = (float*)d_out;

    // ---- workspace layout ----
    char* ws = (char*)d_ws;
    size_t off = 0;
    auto take = [&](size_t bytes) { size_t o = off; off = (off + bytes + 255) & ~(size_t)255; return o; };
    size_t oStart = take((size_t)B * 4);
    size_t oEnd   = take((size_t)B * 4);
    size_t oCent  = take((size_t)B * 12);
    size_t oWq    = take((size_t)H * H * 2);
    size_t oWk    = take((size_t)H * Haug * 2);
    size_t oWv    = take((size_t)H * Haug * 2);
    size_t oWc1   = take((size_t)H * H * 2);
    size_t oWc2   = take((size_t)H * H * 2);
    size_t oWf1   = take((size_t)F * H * 2);
    size_t oWf2   = take((size_t)H * F * 2);
    size_t oBfB   = take((size_t)B * H * 2);      // block_features bf16
    size_t oAug   = take((size_t)N * Haug * 2);   // augmented atoms bf16
    size_t oQ     = take((size_t)B * H * 4);      // Q f32
    size_t oK     = take((size_t)N * 1024);       // Kmat f32 [N,256]; later FFN hidden bf16 [N,512]
    size_t oV     = take((size_t)N * 1024);       // Vmat f32 [N,256]; later FFN out f32 [N,256]
    size_t oLog   = take((size_t)N * 4);
    size_t oCtx   = take((size_t)B * H * 4);
    size_t oCtxB  = take((size_t)B * H * 2);
    size_t oH1    = take((size_t)B * H * 2);
    size_t oUpd   = take((size_t)B * H * 4);
    size_t oX     = take((size_t)N * H * 4);      // post-LN1 f32
    size_t oXB    = take((size_t)N * H * 2);      // post-LN1 bf16
    (void)ws_size;

    int*   pStart = (int*)(ws + oStart);
    int*   pEnd   = (int*)(ws + oEnd);
    float* pCent  = (float*)(ws + oCent);
    u16*   pWqT   = (u16*)(ws + oWq);
    u16*   pWkT   = (u16*)(ws + oWk);
    u16*   pWvT   = (u16*)(ws + oWv);
    u16*   pWc1T  = (u16*)(ws + oWc1);
    u16*   pWc2T  = (u16*)(ws + oWc2);
    u16*   pWf1T  = (u16*)(ws + oWf1);
    u16*   pWf2T  = (u16*)(ws + oWf2);
    u16*   pBfB   = (u16*)(ws + oBfB);
    u16*   pAug   = (u16*)(ws + oAug);
    float* pQ     = (float*)(ws + oQ);
    float* pK     = (float*)(ws + oK);
    float* pV     = (float*)(ws + oV);
    float* pLog   = (float*)(ws + oLog);
    float* pCtx   = (float*)(ws + oCtx);
    u16*   pCtxB  = (u16*)(ws + oCtxB);
    u16*   pH1    = (u16*)(ws + oH1);
    float* pUpd   = (float*)(ws + oUpd);
    float* pX     = (float*)(ws + oX);
    u16*   pXB    = (u16*)(ws + oXB);
    u16*   pFFh   = (u16*)(ws + oK);              // alias: FFN hidden bf16 [N,512]
    float* pFF    = (float*)(ws + oV);            // alias: FFN out f32 [N,256]

    const int TB = 256;                            // 8 waves / workgroup
    auto wgrid = [&](int items) { return dim3((items + 7) / 8); };

    // empty blocks must see start==end==0
    hipMemsetAsync(ws + oStart, 0, (size_t)B * 8, stream);

    // 1) segment boundaries + centroids
    k_boundaries<<<dim3((N + TB - 1) / TB), TB, 0, stream>>>(bid, pStart, pEnd, N);
    k_centroid<<<wgrid(B), TB, 0, stream>>>(pos, pStart, pEnd, pCent, B);

    // 2) weight conversion (f32 [K,N] -> bf16 [N,K])
    auto conv = [&](const float* W, u16* Wt, int K, int Nd) {
        int n = K * Nd;
        k_wt_transpose<<<dim3((n + TB - 1) / TB), TB, 0, stream>>>(W, Wt, K, Nd);
    };
    conv(Wq, pWqT, H, H);
    conv(Wk, pWkT, Haug, H);
    conv(Wv, pWvT, Haug, H);
    conv(Wc1, pWc1T, H, H);
    conv(Wc2, pWc2T, H, H);
    conv(Wf1, pWf1T, H, F);
    conv(Wf2, pWf2T, F, H);
    {
        size_t n = (size_t)B * H;
        k_f2bf<<<dim3((unsigned)((n + TB - 1) / TB)), TB, 0, stream>>>(bfeat, pBfB, n);
    }

    // 3) RBF geometry + augmented bf16 activations
    k_geom_aug<<<wgrid(N), TB, 0, stream>>>(af, pos, bid, pCent, ctr, wid, Wg, bg,
                                            pAug, N, H, R, Hq);

    // 4) projections (WMMA bf16); grid.x covers Nd/64 (16x64 strip per wave)
    k_gemm_bf16<<<dim3(H / 64, (B + 127) / 128), TB, 0, stream>>>(
        pBfB, pWqT, bq, pQ, nullptr, B, H, H, 0);
    k_gemm_bf16<<<dim3(H / 64, (N + 127) / 128), TB, 0, stream>>>(
        pAug, pWkT, bk, pK, nullptr, N, Haug, H, 0);
    k_gemm_bf16<<<dim3(H / 64, (N + 127) / 128), TB, 0, stream>>>(
        pAug, pWvT, bv, pV, nullptr, N, Haug, H, 0);

    // 5) per-atom logits, per-block softmax + context
    k_logits<<<wgrid(N), TB, 0, stream>>>(pQ, pK, bid, pLog, N, H, 1.0f / 16.0f);
    k_attention<<<wgrid(B), TB, 0, stream>>>(pLog, pV, pStart, pEnd, pCtx, B, H);

    // 6) block update MLP (WMMA bf16)
    {
        size_t n = (size_t)B * H;
        k_f2bf<<<dim3((unsigned)((n + TB - 1) / TB)), TB, 0, stream>>>(pCtx, pCtxB, n);
    }
    k_gemm_bf16<<<dim3(H / 64, (B + 127) / 128), TB, 0, stream>>>(
        pCtxB, pWc1T, bc1, nullptr, pH1, B, H, H, 1);
    k_gemm_bf16<<<dim3(H / 64, (B + 127) / 128), TB, 0, stream>>>(
        pH1, pWc2T, bc2, pUpd, nullptr, B, H, H, 0);

    // 7) residual + LN1 (emit f32 + bf16 for FFN input)
    k_add_ln<<<wgrid(N), TB, 0, stream>>>(af, pUpd, bid, g1, b1, pX, pXB, N, H);

    // 8) FFN (WMMA bf16) + residual + LN2 -> output
    k_gemm_bf16<<<dim3(F / 64, (N + 127) / 128), TB, 0, stream>>>(
        pXB, pWf1T, bf1, nullptr, pFFh, N, H, F, 1);
    k_gemm_bf16<<<dim3(H / 64, (N + 127) / 128), TB, 0, stream>>>(
        pFFh, pWf2T, bf2, pFF, nullptr, N, F, H, 0);
    k_add_ln<<<wgrid(N), TB, 0, stream>>>(pX, pFF, nullptr, g2, b2, out, nullptr, N, H);
}
